// PairformerStack_5196910428484
// MI455X (gfx1250) — matrix operand order repaired and checked by reading, not compiled, not run on hardware
//
#include <hip/hip_runtime.h>

typedef __bf16 bf16;
typedef __attribute__((ext_vector_type(16))) __bf16 v16bf;
typedef __attribute__((ext_vector_type(8)))  float  v8f;

#define NSEQ 256
#define NNL  65536L
#define CZ   128
#define CS   384

__device__ __forceinline__ float sigm_f(float x) { return 1.0f / (1.0f + __expf(-x)); }

// ======================= LayerNorm (arbitrary stride) =======================
__global__ __launch_bounds__(256)
void pf_ln_kernel(const float* __restrict__ x, long rowPitch, long stride,
                  const float* __restrict__ g, const float* __restrict__ b,
                  float* __restrict__ outf, bf16* __restrict__ outb,
                  long outPitch, long R, int L)
{
    int wave = threadIdx.x >> 5;
    int l = threadIdx.x & 31;
    long row = (long)blockIdx.x * 8 + wave;
    if (row >= R) return;
    const float* xr = x + row * rowPitch;
    float s = 0.f, s2 = 0.f;
    for (int c = l; c < L; c += 32) {
        float v = xr[(long)c * stride];
        s += v; s2 += v * v;
    }
#pragma unroll
    for (int off = 16; off > 0; off >>= 1) {
        s  += __shfl_xor(s,  off, 32);
        s2 += __shfl_xor(s2, off, 32);
    }
    float mean = s / (float)L;
    float inv  = rsqrtf(s2 / (float)L - mean * mean + 1e-5f);
    for (int c = l; c < L; c += 32) {
        float v = (xr[(long)c * stride] - mean) * inv;
        if (g) v = v * g[c] + b[c];
        if (outf) outf[row * outPitch + c] = v;
        if (outb) outb[row * outPitch + c] = (bf16)v;
    }
}

// ============== weight f32 [K,N] -> bf16 [N,K] (WMMA B-operand) =============
__global__ __launch_bounds__(256)
void pf_wt_kernel(const float* __restrict__ W, bf16* __restrict__ Bt, int K, int N)
{
    long t = (long)blockIdx.x * 256 + threadIdx.x;
    if (t >= (long)K * N) return;
    long n = t / K, k2 = t % K;
    Bt[t] = (bf16)W[k2 * N + n];
}

// ======= pack [i,k,c] (or transpose [k,i,c]) f32 -> channel-major bf16 =======
__global__ __launch_bounds__(256)
void pf_pack_kernel(const float* __restrict__ in, bf16* __restrict__ out, int trans)
{
    long t = (long)blockIdx.x * 256 + threadIdx.x;
    if (t >= NNL * CZ) return;
    long c = t / NNL, r = t % NNL;
    long i = r / NSEQ, k2 = r % NSEQ;
    float v = trans ? in[(k2 * NSEQ + i) * CZ + c] : in[r * CZ + c];
    out[c * NNL + r] = (bf16)v;
}

// ================== transpose z[i,j,:] -> z[j,i,:] (float4) =================
__global__ __launch_bounds__(256)
void pf_trch_kernel(const float4* __restrict__ in, float4* __restrict__ out)
{
    long t = (long)blockIdx.x * 256 + threadIdx.x;  // NN * 32 elements
    long r = t >> 5, c4 = t & 31;
    long i = r / NSEQ, j = r % NSEQ;
    out[(j * NSEQ + i) * 32 + c4] = in[t];
}

__global__ __launch_bounds__(256)
void pf_tr2d_kernel(const float* __restrict__ in, float* __restrict__ out)
{
    int t = blockIdx.x * 256 + threadIdx.x;
    int i = t / NSEQ, j = t % NSEQ;
    out[(long)j * NSEQ + i] = in[t];
}

// === triangle/pair bias: out[h][row] = sum_c x[row,c]*w[c,h]  (row = q*N+k) ==
__global__ __launch_bounds__(256)
void pf_biasdot_kernel(const bf16* __restrict__ x, const float* __restrict__ w,
                       float* __restrict__ out, int H)
{
    long r = (long)blockIdx.x * 256 + threadIdx.x;
    if (r >= NNL) return;
    float acc[16];
    for (int h = 0; h < 16; ++h) acc[h] = 0.f;
    for (int c = 0; c < CZ; ++c) {
        float xv = (float)x[r * CZ + c];
        for (int h = 0; h < H; ++h) acc[h] += xv * w[c * H + h];
    }
    for (int h = 0; h < H; ++h) out[(long)h * NNL + r] = acc[h];
}

// =========================== WMMA bf16 GEMM core ============================
// Register-blocked: each wave computes a 16(M)x64(N) tile (4 accumulators),
// amortizing the A fragment 4x. 4 waves per block split a 64-row stripe.
// epi: 0 plain(+bias1)  1 sigmoid(acc1+b1)  2 sig(acc1+b1)*(acc2+b2)*mask[row]
//      3 silu(acc1)*acc2  4 sig(acc1+b1)*aux (use accum=1)  5 sig(acc1+b1)*aux+acc2
template<int DUAL>
__global__ __launch_bounds__(128)
void pf_gemm_kernel(const bf16* __restrict__ A, long lda,
                    const bf16* __restrict__ B1, const bf16* __restrict__ B2, long ldb,
                    const float* __restrict__ bias1, const float* __restrict__ bias2,
                    const float* __restrict__ aux, const float* __restrict__ mrow,
                    float* __restrict__ dstf, bf16* __restrict__ dstb, long ldd,
                    int M, int N, int K, int epi, int accum, int pad24,
                    long strideA, long strideB, long strideD)
{
    int wave = threadIdx.x >> 5;
    int l    = threadIdx.x & 31;
    int hh   = l >> 4;
    int m    = l & 15;
    int n0   = blockIdx.x * 64;              // block covers 64 output columns
    int m0   = blockIdx.y * 64 + wave * 16;  // 4 waves split 64 rows
    long zb  = blockIdx.z;
    if (m0 >= M) return;

    const bf16* Ar = A + zb * strideA + (long)(m0 + m) * lda;
    const bf16* Br1[4];
    const bf16* Br2[4];
#pragma unroll
    for (int t = 0; t < 4; ++t) {
        Br1[t] = B1 + zb * strideB + (long)(n0 + t * 16 + m) * ldb;
        if (DUAL) Br2[t] = B2 + zb * strideB + (long)(n0 + t * 16 + m) * ldb;
    }

    v8f acc1[4], acc2[4];
#pragma unroll
    for (int t = 0; t < 4; ++t) {
        acc1[t] = (v8f){0.f,0.f,0.f,0.f,0.f,0.f,0.f,0.f};
        acc2[t] = (v8f){0.f,0.f,0.f,0.f,0.f,0.f,0.f,0.f};
    }

    for (int k0 = 0; k0 < K; k0 += 32) {
        v16bf a;
        {   // A frag: lane=row; halves own K {0..15}/{16..31}
            const bf16* ap = Ar + k0 + 16 * hh;
#pragma unroll
            for (int e = 0; e < 16; ++e) a[e] = ap[e];
        }
        __builtin_prefetch((const void*)(Ar + k0 + 128), 0, 0);
        __builtin_prefetch((const void*)(Br1[0] + k0 + 128), 0, 0);
#pragma unroll
        for (int t = 0; t < 4; ++t) {
            v16bf b1v;
            const bf16* bp = Br1[t] + k0 + 16 * hh;
#pragma unroll
            for (int e = 0; e < 16; ++e) b1v[e] = bp[e];
            acc1[t] = __builtin_amdgcn_wmma_f32_16x16x32_bf16(false, a, false, b1v, (short)0, acc1[t], false, false);
            if (DUAL) {
                v16bf b2v;
                const bf16* bp2 = Br2[t] + k0 + 16 * hh;
#pragma unroll
                for (int e = 0; e < 16; ++e) b2v[e] = bp2[e];
                acc2[t] = __builtin_amdgcn_wmma_f32_16x16x32_bf16(false, a, false, b2v, (short)0, acc2[t], false, false);
            }
        }
    }

#pragma unroll
    for (int t = 0; t < 4; ++t) {
        int col = n0 + t * 16 + m;
        float bb1 = bias1 ? bias1[col] : 0.f;
        float bb2 = bias2 ? bias2[col] : 0.f;
        long oc = col;
        if (pad24) oc = (long)(col / 24) * 32 + (col % 24);
#pragma unroll
        for (int r = 0; r < 8; ++r) {
            int row = m0 + r + 8 * hh;
            float v1 = acc1[t][r] + bb1;
            float v2 = acc2[t][r] + bb2;
            float val = v1;
            if (epi == 1)      val = sigm_f(v1);
            else if (epi == 2) { val = sigm_f(v1) * v2; if (mrow) val *= mrow[row]; }
            else if (epi == 3) { float a1 = acc1[t][r]; val = a1 * sigm_f(a1) * acc2[t][r]; }
            else if (epi == 4) { val = sigm_f(v1) * aux[zb * strideD + (long)row * ldd + col]; }
            else if (epi == 5) { val = sigm_f(v1) * aux[(long)row * ldd + col] + v2; }
            long idx = zb * strideD + (long)row * ldd + oc;
            if (dstf) dstf[idx] = accum ? (dstf[idx] + val) : val;
            if (dstb) dstb[idx] = (bf16)val;
        }
    }
}

// ================= fused attention (QK^T -> softmax -> PV) ==================
// one wave per (q-tile of 16, head, batch); S in LDS; WMMA for both matmuls
template<int HASMASK>
__global__ __launch_bounds__(32)
void pf_attn_kernel(const bf16* __restrict__ q, const bf16* __restrict__ kk,
                    const bf16* __restrict__ v, long qld,
                    const float* __restrict__ bias, const float* __restrict__ mask,
                    const float* __restrict__ gate, bf16* __restrict__ out,
                    long gld, int dh_real, float scale, long bstride)
{
    __shared__ float Ssm[16][NSEQ];
    __shared__ bf16  Psm[16][NSEQ];
    __shared__ float rsum[16];
    int l  = threadIdx.x;
    int hh = l >> 4;
    int m  = l & 15;
    int h  = blockIdx.y;
    long bz = blockIdx.z;
    int q0 = blockIdx.x * 16;
    const bf16* qb = q  + bz * bstride;
    const bf16* kb = kk + bz * bstride;
    const bf16* vb = v  + bz * bstride;

    v16bf aq;
    {
        const bf16* qr = qb + (long)(q0 + m) * qld + h * 32 + 16 * hh;
#pragma unroll
        for (int e = 0; e < 16; ++e) aq[e] = qr[e];
    }
    for (int kt = 0; kt < NSEQ / 16; ++kt) {
        int kc = kt * 16 + m;
        v16bf bk;
        {
            const bf16* kr = kb + (long)kc * qld + h * 32 + 16 * hh;
#pragma unroll
            for (int e = 0; e < 16; ++e) bk[e] = kr[e];
        }
        v8f acc = {0.f,0.f,0.f,0.f,0.f,0.f,0.f,0.f};
        acc = __builtin_amdgcn_wmma_f32_16x16x32_bf16(false, aq, false, bk, (short)0, acc, false, false);
        // mask term depends only on the k-column: hoist out of the row loop
        float mterm = 0.f;
        if (HASMASK) mterm = (mask[bz * NSEQ + kc] - 1.0f) * 1.0e9f;
        const float* brow = bias + ((long)h * NSEQ + q0) * NSEQ + kc;
#pragma unroll
        for (int r = 0; r < 8; ++r) {
            float sv = acc[r] * scale + brow[(long)(r + 8 * hh) * NSEQ] + mterm;
            Ssm[r + 8 * hh][kc] = sv;
        }
    }
    __syncthreads();
    if (l < 16) {
        float mx = -3.0e38f;
        for (int j = 0; j < NSEQ; ++j) mx = fmaxf(mx, Ssm[l][j]);
        float sm = 0.f;
        for (int j = 0; j < NSEQ; ++j) {
            float p = __expf(Ssm[l][j] - mx);
            sm += p;
            Psm[l][j] = (bf16)p;
        }
        rsum[l] = sm;
    }
    __syncthreads();

    v8f oacc[2];
    oacc[0] = (v8f){0.f,0.f,0.f,0.f,0.f,0.f,0.f,0.f};
    oacc[1] = (v8f){0.f,0.f,0.f,0.f,0.f,0.f,0.f,0.f};
    for (int kt = 0; kt < NSEQ / 32; ++kt) {
        v16bf ap;
        {
            int kb0 = kt * 32 + 16 * hh;
#pragma unroll
            for (int e = 0; e < 16; ++e) ap[e] = Psm[m][kb0 + e];
        }
#pragma unroll
        for (int nt = 0; nt < 2; ++nt) {
            v16bf bv;
#pragma unroll
            for (int e = 0; e < 16; ++e) {
                int kr2 = kt * 32 + e + 16 * hh;
                bv[e] = vb[(long)kr2 * qld + h * 32 + nt * 16 + m];
            }
            oacc[nt] = __builtin_amdgcn_wmma_f32_16x16x32_bf16(false, ap, false, bv, (short)0, oacc[nt], false, false);
        }
    }
#pragma unroll
    for (int nt = 0; nt < 2; ++nt) {
        int d = nt * 16 + m;
        if (d < dh_real) {
#pragma unroll
            for (int r = 0; r < 8; ++r) {
                int qr2 = q0 + r + 8 * hh;
                long idx = bz * (long)NSEQ * gld + (long)qr2 * gld + (long)h * dh_real + d;
                float val = oacc[nt][r] / rsum[r + 8 * hh];
                out[idx] = (bf16)(val * gate[idx]);
            }
        }
    }
}

// ================================= host =====================================
extern "C" void kernel_launch(void* const* d_in, const int* in_sizes, int n_in,
                              void* d_out, int out_size, void* d_ws, size_t ws_size,
                              hipStream_t stream)
{
    (void)n_in; (void)out_size;
    const size_t MB = 1ull << 20;
    if (ws_size < 257 * MB) return;
    char* Wb = (char*)d_ws;
    float* z_buf = (float*)(Wb + 0);            // persistent z (33.5MB)
    float* r1f   = (float*)(Wb + 34 * MB);      // trimul aux t1 / zt for ending-node
    bf16*  x_bf  = (bf16*) (Wb + 68 * MB);      // LN(z) bf16
    float* f32t  = (float*)(Wb + 86 * MB);      // a/b projections, triatt gate
    bf16*  r4a   = (bf16*) (Wb + 120 * MB);     // a-pack | q | transition hidden
    bf16*  r4b   = (bf16*) (Wb + 137 * MB);     // b-pack | k
    bf16*  r4c   = (bf16*) (Wb + 154 * MB);     // v
    float* ypack = (float*)(Wb + 188 * MB);     // einsum out (channel-major)
    bf16*  o_g   = (bf16*) (Wb + 188 * MB);     // gated attention output (reuse)
    bf16*  yln   = (bf16*) (Wb + 222 * MB);     // LN(y) bf16
    float* biasb = (float*)(Wb + 240 * MB);     // attention bias [H][N][N]
    char*  SMB   = Wb + 245 * MB;
    bf16*  wt1   = (bf16*) (Wb + 253 * MB);
    bf16*  wt2   = (bf16*) (Wb + 255 * MB);

    float* s_buf  = (float*)(SMB + 0x000000);
    bf16*  sln_bf = (bf16*) (SMB + 0x060000);
    float* lnna   = (float*)(SMB + 0x090000);
    bf16*  a_bf   = (bf16*) (SMB + 0x0F0000);
    bf16*  qpad   = (bf16*) (SMB + 0x120000);
    bf16*  kpad   = (bf16*) (SMB + 0x160000);
    bf16*  vpad   = (bf16*) (SMB + 0x1A0000);
    float* g_s    = (float*)(SMB + 0x1E0000);
    float* t2     = (float*)(SMB + 0x240000);
    bf16*  oapb   = (bf16*) (SMB + 0x2A0000);
    bf16*  sx_bf  = (bf16*) (SMB + 0x2D0000);
    bf16*  sh_bf  = (bf16*) (SMB + 0x300000);
    float* maskT  = (float*)(SMB + 0x3C0000);

    hipMemcpyAsync(s_buf, d_in[0], (size_t)NSEQ * CS * 4, hipMemcpyDeviceToDevice, stream);
    hipMemcpyAsync(z_buf, d_in[1], (size_t)NNL * CZ * 4, hipMemcpyDeviceToDevice, stream);
    const float* mask = (const float*)d_in[2];
    pf_tr2d_kernel<<<256, 256, 0, stream>>>(mask, maskT);

    int blk = 0;
    auto P = [&](int i) -> const float* {
        return (const float*)d_in[i] + (size_t)blk * (size_t)(in_sizes[i] / 2);
    };
    auto ln = [&](const float* x, long rowPitch, long stride, const float* g, const float* b,
                  float* outf, bf16* outb, long outPitch, long R, int L) {
        pf_ln_kernel<<<(unsigned)((R + 7) / 8), 256, 0, stream>>>(x, rowPitch, stride, g, b, outf, outb, outPitch, R, L);
    };
    auto gemm = [&](const bf16* A, int lda, const float* W1, const float* W2,
                    int K, int N, int M, const float* b1, const float* b2,
                    const float* aux, const float* mrow,
                    float* dstf, bf16* dstb, int ldd, int epi, int accum, int pad24) {
        long e = (long)K * N;
        unsigned gws = (unsigned)((e + 255) / 256);
        pf_wt_kernel<<<gws, 256, 0, stream>>>(W1, wt1, K, N);
        dim3 grid(N / 64, M / 64, 1);
        if (W2) {
            pf_wt_kernel<<<gws, 256, 0, stream>>>(W2, wt2, K, N);
            pf_gemm_kernel<1><<<grid, 128, 0, stream>>>(
                A, (long)lda, wt1, wt2, (long)K,
                b1, b2, aux, mrow, dstf, dstb, (long)ldd, M, N, K, epi, accum, pad24, 0L, 0L, 0L);
        } else {
            pf_gemm_kernel<0><<<grid, 128, 0, stream>>>(
                A, (long)lda, wt1, (const bf16*)0, (long)K,
                b1, b2, aux, mrow, dstf, dstb, (long)ldd, M, N, K, epi, accum, pad24, 0L, 0L, 0L);
        }
    };
    auto trimul = [&](int tb, bool outgoing) {
        ln(z_buf, CZ, 1, P(tb + 0), P(tb + 1), nullptr, x_bf, CZ, NNL, CZ);
        gemm(x_bf, CZ, P(tb + 2), P(tb + 4), CZ, CZ, (int)NNL, P(tb + 3), P(tb + 5),
             nullptr, mask, f32t, nullptr, CZ, 2, 0, 0);
        pf_pack_kernel<<<32768, 256, 0, stream>>>(f32t, r4a, outgoing ? 0 : 1);
        gemm(x_bf, CZ, P(tb + 6), P(tb + 8), CZ, CZ, (int)NNL, P(tb + 7), P(tb + 9),
             nullptr, mask, f32t, nullptr, CZ, 2, 0, 0);
        pf_pack_kernel<<<32768, 256, 0, stream>>>(f32t, r4b, outgoing ? 0 : 1);
        // triangle einsum: 128 channel-batched 256x256x256 A*B^T GEMMs
        pf_gemm_kernel<0><<<dim3(4, 4, 128), 128, 0, stream>>>(
            r4a, 256L, r4b, (const bf16*)0, 256L,
            nullptr, nullptr, nullptr, nullptr, ypack, nullptr, 256L,
            256, 256, 256, 0, 0, 0, NNL, NNL, NNL);
        ln(ypack, 1, NNL, P(tb + 10), P(tb + 11), nullptr, yln, CZ, NNL, CZ);
        gemm(yln, CZ, P(tb + 12), nullptr, CZ, CZ, (int)NNL, P(tb + 13), nullptr,
             nullptr, nullptr, r1f, nullptr, CZ, 0, 0, 0);
        gemm(x_bf, CZ, P(tb + 14), nullptr, CZ, CZ, (int)NNL, P(tb + 15), nullptr,
             r1f, nullptr, z_buf, nullptr, CZ, 4, 1, 0);
    };
    auto triatt = [&](int tb, float* zsrc, const float* mptr) {
        ln(zsrc, CZ, 1, P(tb + 0), P(tb + 1), nullptr, x_bf, CZ, NNL, CZ);
        gemm(x_bf, CZ, P(tb + 2), nullptr, CZ, CZ, (int)NNL, nullptr, nullptr, nullptr, nullptr,
             nullptr, r4a, CZ, 0, 0, 0);   // q
        gemm(x_bf, CZ, P(tb + 3), nullptr, CZ, CZ, (int)NNL, nullptr, nullptr, nullptr, nullptr,
             nullptr, r4b, CZ, 0, 0, 0);   // k
        gemm(x_bf, CZ, P(tb + 4), nullptr, CZ, CZ, (int)NNL, nullptr, nullptr, nullptr, nullptr,
             nullptr, r4c, CZ, 0, 0, 0);   // v
        gemm(x_bf, CZ, P(tb + 6), nullptr, CZ, CZ, (int)NNL, P(tb + 7), nullptr, nullptr, nullptr,
             f32t, nullptr, CZ, 1, 0, 0);  // gate = sigmoid(x@w_g + b_g)
        pf_biasdot_kernel<<<256, 256, 0, stream>>>(x_bf, P(tb + 5), biasb, 4);
        pf_attn_kernel<1><<<dim3(16, 4, 256), 32, 0, stream>>>(
            r4a, r4b, r4c, 128L, biasb, mptr, f32t, o_g, 128L, 32,
            0.17677669529663687f, (long)NSEQ * CZ);
        gemm(o_g, CZ, P(tb + 8), nullptr, CZ, CZ, (int)NNL, P(tb + 9), nullptr, nullptr, nullptr,
             zsrc, nullptr, CZ, 0, 1, 0);
    };

    for (blk = 0; blk < 2; ++blk) {
        trimul(3, true);
        trimul(19, false);
        triatt(35, z_buf, mask);
        pf_trch_kernel<<<8192, 256, 0, stream>>>((const float4*)z_buf, (float4*)r1f);
        triatt(45, r1f, maskT);
        pf_trch_kernel<<<8192, 256, 0, stream>>>((const float4*)r1f, (float4*)z_buf);
        // pair transition (SwiGLU n=4)
        ln(z_buf, CZ, 1, P(55), P(56), nullptr, x_bf, CZ, NNL, CZ);
        gemm(x_bf, CZ, P(57), P(58), CZ, 512, (int)NNL, nullptr, nullptr, nullptr, nullptr,
             nullptr, r4a, 512, 3, 0, 0);
        gemm(r4a, 512, P(59), nullptr, 512, CZ, (int)NNL, nullptr, nullptr, nullptr, nullptr,
             z_buf, nullptr, CZ, 0, 1, 0);
        // AttentionPairBias with AdaLN
        int ab = 60;
        ln(s_buf, CS, 1, P(ab + 0), P(ab + 1), nullptr, sln_bf, CS, NSEQ, CS);
        ln(s_buf, CS, 1, nullptr, nullptr, lnna, nullptr, CS, NSEQ, CS);
        gemm(sln_bf, CS, P(ab + 2), P(ab + 4), CS, CS, NSEQ, P(ab + 3), nullptr, lnna, nullptr,
             nullptr, a_bf, CS, 5, 0, 0);
        hipMemsetAsync(qpad, 0, (size_t)NSEQ * 512 * 2, stream);
        hipMemsetAsync(kpad, 0, (size_t)NSEQ * 512 * 2, stream);
        hipMemsetAsync(vpad, 0, (size_t)NSEQ * 512 * 2, stream);
        gemm(a_bf, CS, P(ab + 5), nullptr, CS, CS, NSEQ, P(ab + 6), nullptr, nullptr, nullptr,
             nullptr, qpad, 512, 0, 0, 1);
        gemm(a_bf, CS, P(ab + 7), nullptr, CS, CS, NSEQ, nullptr, nullptr, nullptr, nullptr,
             nullptr, kpad, 512, 0, 0, 1);
        gemm(a_bf, CS, P(ab + 8), nullptr, CS, CS, NSEQ, nullptr, nullptr, nullptr, nullptr,
             nullptr, vpad, 512, 0, 0, 1);
        gemm(a_bf, CS, P(ab + 9), nullptr, CS, CS, NSEQ, nullptr, nullptr, nullptr, nullptr,
             g_s, nullptr, CS, 1, 0, 0);
        ln(z_buf, CZ, 1, P(ab + 10), P(ab + 11), nullptr, x_bf, CZ, NNL, CZ);
        pf_biasdot_kernel<<<256, 256, 0, stream>>>(x_bf, P(ab + 12), biasb, 16);
        pf_attn_kernel<0><<<dim3(16, 16, 1), 32, 0, stream>>>(
            qpad, kpad, vpad, 512L, biasb, nullptr, g_s, oapb, (long)CS, 24,
            0.20412414523193154f, 0L);
        gemm(oapb, CS, P(ab + 13), nullptr, CS, CS, NSEQ, nullptr, nullptr, nullptr, nullptr,
             t2, nullptr, CS, 0, 0, 0);
        gemm(sln_bf, CS, P(ab + 14), nullptr, CS, CS, NSEQ, P(ab + 15), nullptr, t2, nullptr,
             s_buf, nullptr, CS, 4, 1, 0);
        // single transition
        int stb = 76;
        ln(s_buf, CS, 1, P(stb + 0), P(stb + 1), nullptr, sx_bf, CS, NSEQ, CS);
        gemm(sx_bf, CS, P(stb + 2), P(stb + 3), CS, 1536, NSEQ, nullptr, nullptr, nullptr, nullptr,
             nullptr, sh_bf, 1536, 3, 0, 0);
        gemm(sh_bf, 1536, P(stb + 4), nullptr, 1536, CS, NSEQ, nullptr, nullptr, nullptr, nullptr,
             s_buf, nullptr, CS, 0, 1, 0);
    }

    float* out = (float*)d_out;
    hipMemcpyAsync(out, s_buf, (size_t)NSEQ * CS * 4, hipMemcpyDeviceToDevice, stream);
    hipMemcpyAsync(out + (size_t)NSEQ * CS, z_buf, (size_t)NNL * CZ * 4, hipMemcpyDeviceToDevice, stream);
}
